// SpatialTemporalDecoder2_17781164605595
// MI455X (gfx1250) — matrix operand order repaired and checked by reading, not compiled, not run on hardware
//
#include <hip/hip_runtime.h>

// ---------------- problem constants ----------------
#define DD   1024
#define SS   512
#define BB   32
#define MR   (BB * SS)          // 16384 rows
#define EPSL 1e-05f

// ---------------- WMMA types ----------------
typedef __attribute__((ext_vector_type(16))) __bf16        bf16x16;
typedef __attribute__((ext_vector_type(8)))  float         f32x8;
typedef __attribute__((ext_vector_type(8)))  unsigned int  u32x8;

__device__ __forceinline__ unsigned short f2bf(float f) {
    unsigned int u = __float_as_uint(f);
    u += 0x7FFFu + ((u >> 16) & 1u);      // round-to-nearest-even
    return (unsigned short)(u >> 16);
}

// 16 contiguous bf16 at p  (A-fragment chunk pattern)
__device__ __forceinline__ bf16x16 frag_from_2x16B(const unsigned short* p) {
    uint4 a = *(const uint4*)p;
    uint4 b = *(const uint4*)(p + 8);
    u32x8 v = {a.x, a.y, a.z, a.w, b.x, b.y, b.z, b.w};
    return __builtin_bit_cast(bf16x16, v);
}

// 16 contiguous bf16 at p  (B-fragment, transposed-weight convention)
__device__ __forceinline__ bf16x16 frag_contig16(const unsigned short* p) {
    uint4 a = *(const uint4*)p;
    uint4 b = *(const uint4*)(p + 8);
    u32x8 v = {a.x, a.y, a.z, a.w, b.x, b.y, b.z, b.w};
    return __builtin_bit_cast(bf16x16, v);
}

// A fragment, 16x32 bf16 tile from row-major [M][K]
__device__ __forceinline__ bf16x16 loadA(const unsigned short* Ab, int lda,
                                         int mBase, int kBase, int lane) {
    int m   = mBase + (lane & 15);
    int off = (lane >> 4) * 16;           // lanes 16-31 start at K+16
    return frag_from_2x16B(Ab + (size_t)m * lda + kBase + off);
}

// B fragment for C = A x W^T with W row-major [N][K] -> contiguous per lane
__device__ __forceinline__ bf16x16 loadBt(const unsigned short* Wb, int ldb,
                                          int nBase, int kBase, int lane) {
    int n = nBase + (lane & 15);
    int g = lane >> 4;                    // lanes 16-31 hold K=16..31
    return frag_contig16(Wb + (size_t)n * ldb + kBase + g * 16);
}

// B fragment for "normal" layout V[K][N] row-major (strided gather)
__device__ __forceinline__ bf16x16 loadBn(const unsigned short* Vb, int ldb,
                                          int nBase, int kBase, int lane) {
    int n  = nBase + (lane & 15);
    int k0 = kBase + (lane >> 4) * 16;
    u32x8 v;
#pragma unroll
    for (int i = 0; i < 8; ++i) {
        unsigned int lo = Vb[(size_t)(k0 + 2 * i)     * ldb + n];
        unsigned int hi = Vb[(size_t)(k0 + 2 * i + 1) * ldb + n];
        v[i] = lo | (hi << 16);
    }
    return __builtin_bit_cast(bf16x16, v);
}

// A fragment converted on the fly from f32 (attention probabilities)
__device__ __forceinline__ bf16x16 loadA_f32(const float* Ab, int lda,
                                             int mBase, int kBase, int lane) {
    int m   = mBase + (lane & 15);
    int off = (lane >> 4) * 16;
    const float* p = Ab + (size_t)m * lda + kBase + off;
    float4 f0 = *(const float4*)(p);
    float4 f1 = *(const float4*)(p + 4);
    float4 f2 = *(const float4*)(p + 8);
    float4 f3 = *(const float4*)(p + 12);
    float fs[16] = {f0.x, f0.y, f0.z, f0.w, f1.x, f1.y, f1.z, f1.w,
                    f2.x, f2.y, f2.z, f2.w, f3.x, f3.y, f3.z, f3.w};
    u32x8 v;
#pragma unroll
    for (int i = 0; i < 8; ++i)
        v[i] = (unsigned int)f2bf(fs[2 * i]) | ((unsigned int)f2bf(fs[2 * i + 1]) << 16);
    return __builtin_bit_cast(bf16x16, v);
}

__device__ __forceinline__ f32x8 wmma_bf16(bf16x16 a, bf16x16 b, f32x8 c) {
    return __builtin_amdgcn_wmma_f32_16x16x32_bf16(false, a, false, b,
                                                   (short)0, c, false, false);
}

// ---------------- conversion ----------------
__global__ __launch_bounds__(256) void cvt_f32_bf16(const float* in,
                                                    unsigned short* out, int n) {
    int i = blockIdx.x * 256 + threadIdx.x;
    if (i < n) out[i] = f2bf(in[i]);
}

// ------- fused GLU dual-GEMM, 32x32 register tile per wave (8 WMMA/k-step) ----
// H = sigmoid(A W1^T + b1) * (A W2^T + b2) + Y
__global__ __launch_bounds__(256) void glu_gemm(const unsigned short* A,
                                                const unsigned short* W1,
                                                const unsigned short* W2,
                                                const float* b1, const float* b2,
                                                const float* Y, float* H,
                                                int Mrows, int N, int K) {
    int lane = threadIdx.x & 31;
    int wid  = (blockIdx.x * 256 + threadIdx.x) >> 5;
    int ntiles32 = N >> 5;
    int mBase = (wid / ntiles32) << 5;
    int nBase = (wid % ntiles32) << 5;
    if (mBase >= Mrows) return;
    f32x8 acc1[2][2] = {}, acc2[2][2] = {};
    for (int k = 0; k < K; k += 32) {
        // prefetch next A chunk (speculative, global_prefetch_b8)
        __builtin_prefetch(A + (size_t)(mBase + (lane & 15)) * K + k + 256, 0, 3);
        bf16x16 a0 = loadA(A, K, mBase, k, lane);
        bf16x16 a1 = loadA(A, K, mBase + 16, k, lane);
        bf16x16 w10 = loadBt(W1, K, nBase, k, lane);
        bf16x16 w11 = loadBt(W1, K, nBase + 16, k, lane);
        bf16x16 w20 = loadBt(W2, K, nBase, k, lane);
        bf16x16 w21 = loadBt(W2, K, nBase + 16, k, lane);
        acc1[0][0] = wmma_bf16(a0, w10, acc1[0][0]);
        acc1[0][1] = wmma_bf16(a0, w11, acc1[0][1]);
        acc1[1][0] = wmma_bf16(a1, w10, acc1[1][0]);
        acc1[1][1] = wmma_bf16(a1, w11, acc1[1][1]);
        acc2[0][0] = wmma_bf16(a0, w20, acc2[0][0]);
        acc2[0][1] = wmma_bf16(a0, w21, acc2[0][1]);
        acc2[1][0] = wmma_bf16(a1, w20, acc2[1][0]);
        acc2[1][1] = wmma_bf16(a1, w21, acc2[1][1]);
    }
#pragma unroll
    for (int mi = 0; mi < 2; ++mi) {
#pragma unroll
        for (int ni = 0; ni < 2; ++ni) {
            int n    = nBase + ni * 16 + (lane & 15);
            int mOff = mBase + mi * 16 + ((lane >> 4) << 3);
            float bias1 = b1[n], bias2 = b2[n];
#pragma unroll
            for (int r = 0; r < 8; ++r) {
                size_t idx = (size_t)(mOff + r) * N + n;
                float s   = acc1[mi][ni][r] + bias1;
                float sig = 1.0f / (1.0f + __expf(-s));
                H[idx] = sig * (acc2[mi][ni][r] + bias2) + Y[idx];
            }
        }
    }
}

// ------- QKV triple GEMM, 2 M-tiles per wave (6 WMMA/k-step) ----------------
__global__ __launch_bounds__(256) void qkv_gemm(const unsigned short* A,
                                                const unsigned short* Wqkv,
                                                const float* bqkv,
                                                unsigned short* Q,
                                                unsigned short* Ko,
                                                unsigned short* V) {
    int lane = threadIdx.x & 31;
    int wid  = (blockIdx.x * 256 + threadIdx.x) >> 5;
    int ntiles = DD >> 4;                 // 64
    int mBase  = (wid / ntiles) << 5;     // 2 m-tiles per wave
    int nBase  = (wid % ntiles) << 4;
    if (mBase >= MR) return;
    const unsigned short* Wq = Wqkv;
    const unsigned short* Wk = Wqkv + (size_t)DD * DD;
    const unsigned short* Wv = Wqkv + 2 * (size_t)DD * DD;
    f32x8 aq[2] = {}, ak[2] = {}, av[2] = {};
    for (int k = 0; k < DD; k += 32) {
        __builtin_prefetch(A + (size_t)(mBase + (lane & 15)) * DD + k + 256, 0, 3);
        bf16x16 a0 = loadA(A, DD, mBase, k, lane);
        bf16x16 a1 = loadA(A, DD, mBase + 16, k, lane);
        bf16x16 bq = loadBt(Wq, DD, nBase, k, lane);
        bf16x16 bk = loadBt(Wk, DD, nBase, k, lane);
        bf16x16 bv = loadBt(Wv, DD, nBase, k, lane);
        aq[0] = wmma_bf16(a0, bq, aq[0]);
        aq[1] = wmma_bf16(a1, bq, aq[1]);
        ak[0] = wmma_bf16(a0, bk, ak[0]);
        ak[1] = wmma_bf16(a1, bk, ak[1]);
        av[0] = wmma_bf16(a0, bv, av[0]);
        av[1] = wmma_bf16(a1, bv, av[1]);
    }
    int n = nBase + (lane & 15);
    float biq = bqkv[n], bik = bqkv[DD + n], biv = bqkv[2 * DD + n];
#pragma unroll
    for (int mi = 0; mi < 2; ++mi) {
        int mOff = mBase + mi * 16 + ((lane >> 4) << 3);
#pragma unroll
        for (int r = 0; r < 8; ++r) {
            size_t idx = (size_t)(mOff + r) * DD + n;
            Q[idx]  = f2bf(aq[mi][r] + biq);
            Ko[idx] = f2bf(ak[mi][r] + bik);
            V[idx]  = f2bf(av[mi][r] + biv);
        }
    }
}

// ---------------- attention scores (causal tiles only) ----------------
__global__ __launch_bounds__(32) void scores_kernel(const unsigned short* Q,
                                                    const unsigned short* Kb,
                                                    float* Attn, float scale) {
    int qt = blockIdx.x, kt = blockIdx.y, b = blockIdx.z;
    if (kt > qt) return;                    // uniform: EXEC stays all-1s
    int lane = threadIdx.x;
    const unsigned short* Qb = Q  + (size_t)b * SS * DD;
    const unsigned short* Kp = Kb + (size_t)b * SS * DD;
    int mBase = qt << 4, nBase = kt << 4;
    f32x8 acc = {};
    for (int k = 0; k < DD; k += 32)
        acc = wmma_bf16(loadA(Qb, DD, mBase, k, lane),
                        loadBt(Kp, DD, nBase, k, lane), acc);
    float* Ab = Attn + (size_t)b * SS * SS;
    int n    = nBase + (lane & 15);
    int mOff = mBase + ((lane >> 4) << 3);
#pragma unroll
    for (int r = 0; r < 8; ++r)
        Ab[(size_t)(mOff + r) * SS + n] = acc[r] * scale;
}

// ---------------- causal softmax (in place, writes zeros above diag) ----------
__global__ __launch_bounds__(512) void softmax_kernel(float* Attn) {
    int q = blockIdx.x, b = blockIdx.y, j = threadIdx.x;
    float* row = Attn + (size_t)b * SS * SS + (size_t)q * SS;
    __shared__ float red[512];
    float v = (j <= q) ? row[j] : -__builtin_inff();
    red[j] = v;
    __syncthreads();
    for (int st = 256; st > 0; st >>= 1) {
        if (j < st) red[j] = fmaxf(red[j], red[j + st]);
        __syncthreads();
    }
    float mx = red[0];
    __syncthreads();
    float e = (j <= q) ? __expf(v - mx) : 0.0f;
    red[j] = e;
    __syncthreads();
    for (int st = 256; st > 0; st >>= 1) {
        if (j < st) red[j] += red[j + st];
        __syncthreads();
    }
    row[j] = e * (1.0f / red[0]);
}

// ---- ctx = attn @ V : one A fragment feeds 2 N-tiles (halves f32->bf16 work) -
__global__ __launch_bounds__(32) void ctx_gemm(const float* Attn,
                                               const unsigned short* V,
                                               unsigned short* Ctx) {
    int mt = blockIdx.x, nt = blockIdx.y, b = blockIdx.z;
    int lane = threadIdx.x;
    const float* Ab          = Attn + (size_t)b * SS * SS;
    const unsigned short* Vb = V    + (size_t)b * SS * DD;
    int mBase = mt << 4, nBase = nt << 5;
    int kEnd = (mBase + 16 + 31) & ~31;     // causal clip (attn is 0 beyond)
    if (kEnd > SS) kEnd = SS;
    f32x8 acc0 = {}, acc1 = {};
    for (int k = 0; k < kEnd; k += 32) {
        bf16x16 a = loadA_f32(Ab, SS, mBase, k, lane);
        acc0 = wmma_bf16(a, loadBn(Vb, DD, nBase, k, lane), acc0);
        acc1 = wmma_bf16(a, loadBn(Vb, DD, nBase + 16, k, lane), acc1);
    }
    int mOff = mBase + ((lane >> 4) << 3);
    unsigned short* Cb = Ctx + (size_t)b * SS * DD;
    int n0 = nBase + (lane & 15);
#pragma unroll
    for (int r = 0; r < 8; ++r) {
        Cb[(size_t)(mOff + r) * DD + n0]      = f2bf(acc0[r]);
        Cb[(size_t)(mOff + r) * DD + n0 + 16] = f2bf(acc1[r]);
    }
}

// ------- generic GEMM (A W^T + b) -> bf16, 32x32 tile per wave --------------
__global__ __launch_bounds__(256) void gemm_bias_bf16(const unsigned short* A,
                                                      const unsigned short* W,
                                                      const float* bias,
                                                      unsigned short* Out,
                                                      int Mrows, int N, int K) {
    int lane = threadIdx.x & 31;
    int wid  = (blockIdx.x * 256 + threadIdx.x) >> 5;
    int ntiles32 = N >> 5;
    int mBase = (wid / ntiles32) << 5;
    int nBase = (wid % ntiles32) << 5;
    if (mBase >= Mrows) return;
    f32x8 acc[2][2] = {};
    for (int k = 0; k < K; k += 32) {
        __builtin_prefetch(A + (size_t)(mBase + (lane & 15)) * K + k + 256, 0, 3);
        bf16x16 a0 = loadA(A, K, mBase, k, lane);
        bf16x16 a1 = loadA(A, K, mBase + 16, k, lane);
        bf16x16 b0 = loadBt(W, K, nBase, k, lane);
        bf16x16 b1f = loadBt(W, K, nBase + 16, k, lane);
        acc[0][0] = wmma_bf16(a0, b0, acc[0][0]);
        acc[0][1] = wmma_bf16(a0, b1f, acc[0][1]);
        acc[1][0] = wmma_bf16(a1, b0, acc[1][0]);
        acc[1][1] = wmma_bf16(a1, b1f, acc[1][1]);
    }
#pragma unroll
    for (int mi = 0; mi < 2; ++mi) {
#pragma unroll
        for (int ni = 0; ni < 2; ++ni) {
            int n    = nBase + ni * 16 + (lane & 15);
            int mOff = mBase + mi * 16 + ((lane >> 4) << 3);
            float bi = bias[n];
#pragma unroll
            for (int r = 0; r < 8; ++r)
                Out[(size_t)(mOff + r) * N + n] = f2bf(acc[mi][ni][r] + bi);
        }
    }
}

// ---------------- LayerNorm over D, optional bf16 copy ----------------
__global__ __launch_bounds__(256) void ln_kernel(const float* H, const float* g,
                                                 const float* beta, float* outF,
                                                 unsigned short* outBf) {
    int row = blockIdx.x, t = threadIdx.x;
    const float* h = H + (size_t)row * DD;
    __shared__ float red[256];
    float4 xv = *(const float4*)(h + t * 4);
    float s = xv.x + xv.y + xv.z + xv.w;
    red[t] = s;
    __syncthreads();
    for (int st = 128; st > 0; st >>= 1) {
        if (t < st) red[t] += red[t + st];
        __syncthreads();
    }
    float mu = red[0] * (1.0f / DD);
    __syncthreads();
    float d0 = xv.x - mu, d1 = xv.y - mu, d2 = xv.z - mu, d3 = xv.w - mu;
    red[t] = d0 * d0 + d1 * d1 + d2 * d2 + d3 * d3;
    __syncthreads();
    for (int st = 128; st > 0; st >>= 1) {
        if (t < st) red[t] += red[t + st];
        __syncthreads();
    }
    float rs = rsqrtf(red[0] * (1.0f / DD) + EPSL);
    float dv[4] = {d0, d1, d2, d3};
#pragma unroll
    for (int i = 0; i < 4; ++i) {
        int c = t * 4 + i;
        float o = dv[i] * rs * g[c] + beta[c];
        size_t idx = (size_t)row * DD + c;
        outF[idx] = o;
        if (outBf) outBf[idx] = f2bf(o);
    }
}

// ---------------- host side ----------------
extern "C" void kernel_launch(void* const* d_in, const int* in_sizes, int n_in,
                              void* d_out, int out_size, void* d_ws, size_t ws_size,
                              hipStream_t stream) {
    const float* X     = (const float*)d_in[0];   // vsn_local_future
    const float* Yg    = (const float*)d_in[1];   // vsn_global
    const float* w1a   = (const float*)d_in[2];
    const float* b1a   = (const float*)d_in[3];
    const float* w2a   = (const float*)d_in[4];
    const float* b2a   = (const float*)d_in[5];
    const float* g1    = (const float*)d_in[6];
    const float* beta1 = (const float*)d_in[7];
    const float* in_w  = (const float*)d_in[8];
    const float* in_b  = (const float*)d_in[9];
    const float* out_w = (const float*)d_in[10];
    const float* out_b = (const float*)d_in[11];
    const float* w1b   = (const float*)d_in[12];
    const float* b1b   = (const float*)d_in[13];
    const float* w2b   = (const float*)d_in[14];
    const float* b2b   = (const float*)d_in[15];
    const float* g2    = (const float*)d_in[16];
    const float* beta2 = (const float*)d_in[17];

    float* glu_delta = (float*)d_out;                     // [MR, DD]
    float* glu_phi   = glu_delta + (size_t)MR * DD;       // [MR, DD]
    float* attn      = glu_phi   + (size_t)MR * DD;       // [BB, SS, SS]

    const size_t MD2 = (size_t)MR * DD * 2;               // 32 MB
    const size_t DD2 = (size_t)DD * DD * 2;               // 2 MB
    char* ws = (char*)d_ws;
    unsigned short* Xbf    = (unsigned short*)(ws);              // also Q bf16
    unsigned short* Kbf    = (unsigned short*)(ws + MD2);        // also attn_out bf16
    unsigned short* Vbf    = (unsigned short*)(ws + 2 * MD2);
    unsigned short* Pbf    = (unsigned short*)(ws + 3 * MD2);    // phi bf16, then ctx bf16
    unsigned short* w1aBf  = (unsigned short*)(ws + 4 * MD2);
    unsigned short* w2aBf  = (unsigned short*)(ws + 4 * MD2 + DD2);
    unsigned short* w1bBf  = (unsigned short*)(ws + 4 * MD2 + 2 * DD2);
    unsigned short* w2bBf  = (unsigned short*)(ws + 4 * MD2 + 3 * DD2);
    unsigned short* outwBf = (unsigned short*)(ws + 4 * MD2 + 4 * DD2);
    unsigned short* inwBf  = (unsigned short*)(ws + 4 * MD2 + 5 * DD2);
    float*          Hbuf   = (float*)(ws + 4 * MD2 + 8 * DD2);   // 64 MB

    const int nAct = MR * DD;        // 16,777,216
    const int nW   = DD * DD;        // 1,048,576
    // 1) conversions to bf16
    cvt_f32_bf16<<<nAct / 256, 256, 0, stream>>>(X, Xbf, nAct);
    cvt_f32_bf16<<<nW / 256, 256, 0, stream>>>(w1a, w1aBf, nW);
    cvt_f32_bf16<<<nW / 256, 256, 0, stream>>>(w2a, w2aBf, nW);
    cvt_f32_bf16<<<3 * nW / 256, 256, 0, stream>>>(in_w, inwBf, 3 * nW);
    cvt_f32_bf16<<<nW / 256, 256, 0, stream>>>(out_w, outwBf, nW);
    cvt_f32_bf16<<<nW / 256, 256, 0, stream>>>(w1b, w1bBf, nW);
    cvt_f32_bf16<<<nW / 256, 256, 0, stream>>>(w2b, w2bBf, nW);

    // 32x32 register-tiled GEMMs: (MR/32)*(DD/32) = 16384 waves -> 2048 blocks
    const int blocks32 = ((MR / 32) * (DD / 32)) / 8;
    // QKV: (MR/32)*(DD/16) = 32768 waves -> 4096 blocks
    const int blocksQKV = ((MR / 32) * (DD / 16)) / 8;

    // 2) GLU #1 -> H
    glu_gemm<<<blocks32, 256, 0, stream>>>(Xbf, w1aBf, w2aBf, b1a, b2a, Yg,
                                           Hbuf, MR, DD, DD);
    // 3) LN #1 -> glu_phi (f32 out) + phi bf16
    ln_kernel<<<MR, 256, 0, stream>>>(Hbuf, g1, beta1, glu_phi, Pbf);
    // 4) QKV (Q reuses X buffer)
    qkv_gemm<<<blocksQKV, 256, 0, stream>>>(Pbf, inwBf, in_b, Xbf, Kbf, Vbf);
    // 5) scores -> attn region of d_out (scale = 1/sqrt(D))
    scores_kernel<<<dim3(SS / 16, SS / 16, BB), 32, 0, stream>>>(Xbf, Kbf, attn,
                                                                 0.03125f);
    // 6) causal softmax in place
    softmax_kernel<<<dim3(SS, BB), 512, 0, stream>>>(attn);
    // 7) ctx = attn @ V -> bf16 (reuses phi buffer)
    ctx_gemm<<<dim3(SS / 16, DD / 32, BB), 32, 0, stream>>>(attn, Vbf, Pbf);
    // 8) output projection -> attn_out bf16 (reuses K buffer)
    gemm_bias_bf16<<<blocks32, 256, 0, stream>>>(Pbf, outwBf, out_b, Kbf,
                                                 MR, DD, DD);
    // 9) GLU #2 (residual = glu_phi) -> H
    glu_gemm<<<blocks32, 256, 0, stream>>>(Kbf, w1bBf, w2bBf, b1b, b2b,
                                           glu_phi, Hbuf, MR, DD, DD);
    // 10) LN #2 -> glu_delta
    ln_kernel<<<MR, 256, 0, stream>>>(Hbuf, g2, beta2, glu_delta, nullptr);
}
